// PointNetPlusPlus_65764539236698
// MI455X (gfx1250) — matrix operand order repaired and checked by reading, not compiled, not run on hardware
//
#include <hip/hip_runtime.h>

// ---------------------------------------------------------------------------
// PointNet++ fused pipeline for gfx1250 (MI455X).
// All MLP Linears use v_wmma_f32_16x16x32_f16 (f16 in / f32 accumulate).
// Weights are pre-packed into WMMA B-fragment order so the GEMM inner loop
// does contiguous b128 loads only. K-reduction uses two independent
// accumulator chains to break WMMA->WMMA RAW latency. Activations stay in LDS.
// ---------------------------------------------------------------------------

typedef __attribute__((ext_vector_type(16))) _Float16 v16h;
typedef __attribute__((ext_vector_type(8)))  _Float16 v8h;
typedef __attribute__((ext_vector_type(8)))  float    v8f;

#define BATCH 16
#define KNBR  25
#define XS    264   // LDS row stride (f16) for SA stages; 264*2 = 528 = 33*16B

// ------------------------------- weight pack -------------------------------
// Pack W[Cin x Cout] (f32) into f16 WMMA-B fragment order, K zero-padded to
// Kpad (multiple of 32):
//   dst[(((chunk*NT + nt)*32 + lane)*16 + i] = W[k, col]
//   k = chunk*32 + (i&7) + (lane>>4)*8 + (i>>3)*16,  col = nt*16 + (lane&15),  NT = Cout/16
__global__ void pack_w_kernel(const float* __restrict__ src, _Float16* __restrict__ dst,
                              int Cin, int Cout, int Kpad) {
  int e = blockIdx.x * blockDim.x + threadIdx.x;
  int tot = Kpad * Cout;
  if (e >= tot) return;
  int NT = Cout >> 4;
  int i    = e & 15;
  int lane = (e >> 4) & 31;
  int t    = e >> 9;           // chunk*NT + nt
  int nt    = t % NT;
  int chunk = t / NT;
  int k   = chunk * 32 + (i & 7) + ((lane >> 4) << 3) + ((i >> 3) << 4);
  int col = nt * 16 + (lane & 15);
  float v = (k < Cin) ? src[(size_t)k * Cout + col] : 0.0f;
  dst[e] = (_Float16)v;
}

// ------------------------------- FPS ---------------------------------------
__global__ __launch_bounds__(256) void fps_kernel(const float* __restrict__ pos_all,
                                                  int N, int M, int* __restrict__ idx_all) {
  __shared__ float dist[4096];
  __shared__ float rv[256];
  __shared__ int   ri[256];
  int b = blockIdx.x, tid = threadIdx.x;
  const float* pos = pos_all + (size_t)b * N * 3;
  int* idx = idx_all + (size_t)b * M;
  for (int i = tid; i < N; i += 256) dist[i] = 1e30f;
  if (tid == 0) idx[0] = 0;
  __syncthreads();
  int last = 0;
  for (int s = 1; s < M; ++s) {
    float lx = pos[last * 3 + 0], ly = pos[last * 3 + 1], lz = pos[last * 3 + 2];
    float bv = -1.0f; int bi = 0x7fffffff;
    for (int i = tid; i < N; i += 256) {
      float dx = pos[i * 3 + 0] - lx;
      float dy = pos[i * 3 + 1] - ly;
      float dz = pos[i * 3 + 2] - lz;
      float d = dx * dx + dy * dy + dz * dz;
      float dd = dist[i];
      if (d < dd) { dd = d; dist[i] = d; }
      if (dd > bv) { bv = dd; bi = i; }   // ascending i keeps first max
    }
    rv[tid] = bv; ri[tid] = bi;
    __syncthreads();
    for (int off = 128; off > 0; off >>= 1) {
      if (tid < off) {
        float ov = rv[tid + off]; int oi = ri[tid + off];
        if (ov > rv[tid] || (ov == rv[tid] && oi < ri[tid])) { rv[tid] = ov; ri[tid] = oi; }
      }
      __syncthreads();
    }
    last = ri[0];
    if (tid == 0) idx[s] = last;
    __syncthreads();
  }
}

// ------------------------------- centroid gather ---------------------------
__global__ void gather_ctr_kernel(const float* __restrict__ pos_all,
                                  const int* __restrict__ idx_all,
                                  float* __restrict__ ctr_all, int N, int M) {
  int t = blockIdx.x * blockDim.x + threadIdx.x;
  if (t >= BATCH * M) return;
  int b = t / M;
  int pi = idx_all[t];
  const float* p = pos_all + ((size_t)b * N + pi) * 3;
  float* c = ctr_all + (size_t)t * 3;
  c[0] = p[0]; c[1] = p[1]; c[2] = p[2];
}

// ------------------------------- ball query --------------------------------
__global__ void ball_query_kernel(const float* __restrict__ pos_all,
                                  const float* __restrict__ ctr_all,
                                  int* __restrict__ nbr_all, int* __restrict__ cnt_all,
                                  int N, int M, float r2) {
  int t = blockIdx.x * blockDim.x + threadIdx.x;
  if (t >= BATCH * M) return;
  int b = t / M;
  const float* pos = pos_all + (size_t)b * N * 3;
  const float* c = ctr_all + (size_t)t * 3;
  int* nbr = nbr_all + (size_t)t * KNBR;
  float cx = c[0], cy = c[1], cz = c[2];
  int cn = 0;
  for (int i = 0; i < N && cn < KNBR; ++i) {
    float dx = pos[i * 3 + 0] - cx;
    float dy = pos[i * 3 + 1] - cy;
    float dz = pos[i * 3 + 2] - cz;
    if (dx * dx + dy * dy + dz * dz <= r2) nbr[cn++] = i;
  }
  cnt_all[t] = cn;
  for (int k = cn; k < KNBR; ++k) nbr[k] = 0;
}

// --------------------------- WMMA GEMM tile core ---------------------------
// Computes one 16x16 output tile accumulated over KPAD with two independent
// WMMA accumulator chains (even/odd k-chunks) to avoid serial D->C RAW stalls.
// A frag (16x32 f16, ISA 7.12.2): lane row = lane&15; halves 0-7 <- K base,
// halves 8-15 <- K base+16; base = kc*32 + 8*(lane>=16). Two aligned 16B loads.
// B frag: packed order, one contiguous 32B load per lane.
template <int KPAD, int NT>
__device__ __forceinline__ v8f wmma_tile(const _Float16* __restrict__ ap,
                                         const _Float16* __restrict__ bp) {
  constexpr int NCHUNK = KPAD >> 5;
  constexpr size_t BSTEP = (size_t)NT * 32 * 16;
  v8f acc0 = {};
  v8f acc1 = {};
  int kc = 0;
#pragma unroll
  for (; kc + 1 < NCHUNK; kc += 2) {
    v16h a0, b0, a1, b1;
    ((v8h*)&a0)[0] = *(const v8h*)(ap);
    ((v8h*)&a0)[1] = *(const v8h*)(ap + 16);
    ((v8h*)&a1)[0] = *(const v8h*)(ap + 32);
    ((v8h*)&a1)[1] = *(const v8h*)(ap + 48);
    b0 = *(const v16h*)(bp);
    b1 = *(const v16h*)(bp + BSTEP);
    acc0 = __builtin_amdgcn_wmma_f32_16x16x32_f16(false, a0, false, b0,
                                                  (short)0, acc0, false, false);
    acc1 = __builtin_amdgcn_wmma_f32_16x16x32_f16(false, a1, false, b1,
                                                  (short)0, acc1, false, false);
    ap += 64;
    bp += 2 * BSTEP;
  }
  if constexpr (NCHUNK & 1) {
    v16h a0, b0;
    ((v8h*)&a0)[0] = *(const v8h*)(ap);
    ((v8h*)&a0)[1] = *(const v8h*)(ap + 16);
    b0 = *(const v16h*)(bp);
    acc0 = __builtin_amdgcn_wmma_f32_16x16x32_f16(false, a0, false, b0,
                                                  (short)0, acc0, false, false);
  }
  if constexpr (NCHUNK > 1) acc0 = acc0 + acc1;
  return acc0;
}

// Y[32 x COUT] = (relu(X[32 x KPAD] @ W + bias)) * scale + shift -> LDS (f16)
template <int KPAD, int COUT, int INS, int OUTS>
__device__ __forceinline__ void mlp_layer_lds(const _Float16* __restrict__ Xin,
                                              _Float16* __restrict__ Xout,
                                              const _Float16* __restrict__ Wp,
                                              const float* __restrict__ bias,
                                              const float* __restrict__ scale,
                                              const float* __restrict__ shift) {
  constexpr int NT = COUT >> 4;
  constexpr int NTILES = NT * 2;
  int lane = threadIdx.x & 31;
  int wave = threadIdx.x >> 5;
  int nwaves = blockDim.x >> 5;
  int hi = lane >> 4, l15 = lane & 15;
  for (int t = wave; t < NTILES; t += nwaves) {
    int mt = t & 1, nt = t >> 1;
    int rowbase = mt * 16, colbase = nt * 16;
    const _Float16* ap = Xin + (size_t)(rowbase + l15) * INS + hi * 8;
    const _Float16* bp = Wp + ((size_t)nt * 32 + lane) * 16;
    v8f acc = wmma_tile<KPAD, NT>(ap, bp);
    int col = colbase + l15;
    float bi = bias[col], sc = scale[col], sh = shift[col];
#pragma unroll
    for (int v = 0; v < 8; ++v) {
      int row = rowbase + v + hi * 8;
      float y = acc[v] + bi;
      y = (y > 0.0f ? y : 0.0f) * sc + sh;
      Xout[(size_t)row * OUTS + col] = (_Float16)y;
    }
  }
}

// Same GEMM, epilogue streams f32 rows straight to global memory.
template <int KPAD, int COUT, int INS>
__device__ __forceinline__ void mlp_layer_glob(const _Float16* __restrict__ Xin,
                                               float* __restrict__ out, int ldo,
                                               int grbase, int Rlim,
                                               const _Float16* __restrict__ Wp,
                                               const float* __restrict__ bias,
                                               const float* __restrict__ scale,
                                               const float* __restrict__ shift) {
  constexpr int NT = COUT >> 4;
  constexpr int NTILES = NT * 2;
  int lane = threadIdx.x & 31;
  int wave = threadIdx.x >> 5;
  int nwaves = blockDim.x >> 5;
  int hi = lane >> 4, l15 = lane & 15;
  for (int t = wave; t < NTILES; t += nwaves) {
    int mt = t & 1, nt = t >> 1;
    int rowbase = mt * 16, colbase = nt * 16;
    const _Float16* ap = Xin + (size_t)(rowbase + l15) * INS + hi * 8;
    const _Float16* bp = Wp + ((size_t)nt * 32 + lane) * 16;
    v8f acc = wmma_tile<KPAD, NT>(ap, bp);
    int col = colbase + l15;
    float bi = bias[col], sc = scale[col], sh = shift[col];
#pragma unroll
    for (int v = 0; v < 8; ++v) {
      int gr = grbase + rowbase + v + hi * 8;
      float y = acc[v] + bi;
      y = (y > 0.0f ? y : 0.0f) * sc + sh;
      if (gr < Rlim) out[(size_t)gr * ldo + col] = y;
    }
  }
}

// ------------------------------- fused SA stage ----------------------------
struct SAArgs {
  const float* feat;     // [B][N][CF] or nullptr
  const float* pos;      // [B][N][3]
  const float* ctr;      // [B][M][3]
  const int*   nbr;      // [B][M][KNBR]
  const int*   cnt;      // [B][M]
  float*       feat_out; // [B][M][C2]
  int N, M;
  const _Float16 *W0, *W1, *W2;
  const float *b0, *s0, *h0, *b1, *s1, *h1, *b2, *s2, *h2;
};

template <int CF, int K0, int C0, int K1, int C1, int K2, int C2>
__global__ __launch_bounds__(128) void sa_stage_kernel(SAArgs a) {
  __shared__ _Float16 Xa[32 * XS];
  __shared__ _Float16 Xb[32 * XS];
  int blk = blockIdx.x;
  int b = blk / a.M, m = blk % a.M;
  int tid = threadIdx.x;
  const int* nb = a.nbr + (size_t)(b * a.M + m) * KNBR;
  const float* c3 = a.ctr + (size_t)(b * a.M + m) * 3;
  constexpr int CIN = CF + 3;

  // Build X0 (rows >= KNBR or cols >= CIN zero-filled)
  for (int e = tid; e < 32 * K0; e += 128) {
    int r = e / K0, c = e % K0;
    _Float16 v = (_Float16)0.0f;
    if (r < KNBR && c < CIN) {
      int pi = nb[r];
      float f;
      if (c < CF) f = a.feat[((size_t)b * a.N + pi) * CF + c];
      else        f = a.pos[((size_t)b * a.N + pi) * 3 + (c - CF)] - c3[c - CF];
      v = (_Float16)f;
    }
    Xa[(size_t)r * XS + c] = v;
  }
  __syncthreads();
  mlp_layer_lds<K0, C0, XS, XS>(Xa, Xb, a.W0, a.b0, a.s0, a.h0);
  __syncthreads();
  mlp_layer_lds<K1, C1, XS, XS>(Xb, Xa, a.W1, a.b1, a.s1, a.h1);
  __syncthreads();
  mlp_layer_lds<K2, C2, XS, XS>(Xa, Xb, a.W2, a.b2, a.s2, a.h2);
  __syncthreads();

  // Masked max-pool over valid rows (slots 0..cnt-1)
  int cn = a.cnt[b * a.M + m];
  for (int c = tid; c < C2; c += 128) {
    float mx = -1e30f;
    for (int r = 0; r < cn; ++r) {
      float v = (float)Xb[(size_t)r * XS + c];
      if (v > mx) mx = v;
    }
    a.feat_out[((size_t)b * a.M + m) * C2 + c] = mx;
  }
}

// ------------------------------- global MLP stage --------------------------
struct GlobArgs {
  const float* feat;   // [R][256]
  const float* pos;    // [R][3]
  float* g;            // [R][1024]
  int R;
  const _Float16 *W0, *W1, *W2;
  const float *b0, *s0, *h0, *b1, *s1, *h1, *b2, *s2, *h2;
};

#define GAS 520   // wide LDS stride (>=512); 520*2 = 1040 = 65*16B
#define GBS 264   // narrow LDS stride (>=256)

__global__ __launch_bounds__(128) void glob_stage_kernel(GlobArgs a) {
  __shared__ _Float16 Xa[32 * GAS];
  __shared__ _Float16 Xb[32 * GBS];
  int r0 = blockIdx.x * 32;
  int tid = threadIdx.x;
  // X0 = [feat(256), pos(3)] padded to 288
  for (int e = tid; e < 32 * 288; e += 128) {
    int r = e / 288, c = e % 288;
    int gr = r0 + r;
    _Float16 v = (_Float16)0.0f;
    if (gr < a.R && c < 259) {
      float f = (c < 256) ? a.feat[(size_t)gr * 256 + c]
                          : a.pos[(size_t)gr * 3 + (c - 256)];
      v = (_Float16)f;
    }
    Xa[(size_t)r * GAS + c] = v;
  }
  __syncthreads();
  mlp_layer_lds<288, 256, GAS, GBS>(Xa, Xb, a.W0, a.b0, a.s0, a.h0);
  __syncthreads();
  mlp_layer_lds<256, 512, GBS, GAS>(Xb, Xa, a.W1, a.b1, a.s1, a.h1);
  __syncthreads();
  mlp_layer_glob<512, 1024, GAS>(Xa, a.g, 1024, r0, a.R, a.W2, a.b2, a.s2, a.h2);
}

// ------------------------------- global max pool ---------------------------
__global__ void pool_kernel(const float* __restrict__ g, float* __restrict__ pooled,
                            _Float16* __restrict__ p16, int Mr, int C) {
  int b = blockIdx.x;
  for (int c = threadIdx.x; c < C; c += blockDim.x) {
    float mx = -1e30f;
    for (int r = 0; r < Mr; ++r) {
      float v = g[((size_t)b * Mr + r) * C + c];
      if (v > mx) mx = v;
    }
    pooled[(size_t)b * C + c] = mx;
    p16[(size_t)b * C + c] = (_Float16)mx;
  }
}

// ------------------------------- final linear ------------------------------
// out[16 x 512] = A[16 x 1024] @ W[1024 x 512] + bias (packed W, single block)
__global__ __launch_bounds__(128) void final_linear_kernel(const _Float16* __restrict__ A,
                                                           const _Float16* __restrict__ Wp,
                                                           const float* __restrict__ bias,
                                                           float* __restrict__ out) {
  constexpr int NT = 32;      // 512/16
  int lane = threadIdx.x & 31;
  int wave = threadIdx.x >> 5;
  int hi = lane >> 4, l15 = lane & 15;
  for (int nt = wave; nt < NT; nt += 4) {
    const _Float16* ap = A + (size_t)l15 * 1024 + hi * 8;
    const _Float16* bp = Wp + ((size_t)nt * 32 + lane) * 16;
    v8f acc = wmma_tile<1024, NT>(ap, bp);
    int col = nt * 16 + l15;
    float bi = bias[col];
#pragma unroll
    for (int v = 0; v < 8; ++v)
      out[(size_t)(v + hi * 8) * 512 + col] = acc[v] + bi;
  }
}

// ---------------------------------------------------------------------------
// Host launcher
// ---------------------------------------------------------------------------
static inline size_t align256(size_t x) { return (x + 255) & ~(size_t)255; }

extern "C" void kernel_launch(void* const* d_in, const int* in_sizes, int n_in,
                              void* d_out, int out_size, void* d_ws, size_t ws_size,
                              hipStream_t stream) {
  (void)in_sizes; (void)n_in; (void)out_size; (void)ws_size;

  // Input order assumes jax pytree flatten (dict keys sorted alphabetically):
  // 0: positions; 1-12: glob (3x [W,b,scale,shift]); 13-14: lin [W,b];
  // 15-26: sa1; 27-38: sa2; 39-50: sa3; 51-62: sa4.
  const float* positions = (const float*)d_in[0];
  const int GLB = 1, LIN_W = 13, LIN_B = 14, SA1 = 15, SA2 = 27, SA3 = 39, SA4 = 51;

  const int M1 = 820, M2 = 205, M3 = 52, M4 = 13;  // ceil(ratio * N) chain
  const float R2[4] = {9.0f, 16.0f, 25.0f, 49.0f};

  // Weight descriptors (d_in index, Cin, Cout, Kpad)
  struct LD { int idx, Cin, Cout, Kpad; };
  const LD L[16] = {
    {SA1 + 0, 3, 64, 32},     {SA1 + 4, 64, 64, 64},     {SA1 + 8, 64, 128, 64},
    {SA2 + 0, 131, 128, 160}, {SA2 + 4, 128, 128, 128},  {SA2 + 8, 128, 128, 128},
    {SA3 + 0, 131, 128, 160}, {SA3 + 4, 128, 128, 128},  {SA3 + 8, 128, 128, 128},
    {SA4 + 0, 131, 128, 160}, {SA4 + 4, 128, 128, 128},  {SA4 + 8, 128, 256, 128},
    {GLB + 0, 259, 256, 288}, {GLB + 4, 256, 512, 256},  {GLB + 8, 512, 1024, 512},
    {LIN_W,   1024, 512, 1024},
  };

  // ---- workspace layout ----
  char* ws = (char*)d_ws;
  size_t off = 0;
  auto alloc = [&](size_t bytes) -> char* {
    char* p = ws + off;
    off = align256(off + bytes);
    return p;
  };
  size_t w16_elems = 0;
  for (int i = 0; i < 16; ++i) w16_elems += (size_t)L[i].Kpad * L[i].Cout;
  _Float16* w16 = (_Float16*)alloc(w16_elems * sizeof(_Float16));
  _Float16* wp[16];
  { size_t wo = 0;
    for (int i = 0; i < 16; ++i) { wp[i] = w16 + wo; wo += (size_t)L[i].Kpad * L[i].Cout; } }

  int*   idx1 = (int*)alloc((size_t)BATCH * M1 * 4);
  float* ctr1 = (float*)alloc((size_t)BATCH * M1 * 3 * 4);
  float* feat1 = (float*)alloc((size_t)BATCH * M1 * 128 * 4);
  int*   idx2 = (int*)alloc((size_t)BATCH * M2 * 4);
  float* ctr2 = (float*)alloc((size_t)BATCH * M2 * 3 * 4);
  float* feat2 = (float*)alloc((size_t)BATCH * M2 * 128 * 4);
  int*   idx3 = (int*)alloc((size_t)BATCH * M3 * 4);
  float* ctr3 = (float*)alloc((size_t)BATCH * M3 * 3 * 4);
  float* feat3 = (float*)alloc((size_t)BATCH * M3 * 128 * 4);
  int*   idx4 = (int*)alloc((size_t)BATCH * M4 * 4);
  float* ctr4 = (float*)alloc((size_t)BATCH * M4 * 3 * 4);
  float* feat4 = (float*)alloc((size_t)BATCH * M4 * 256 * 4);
  int*   nbr  = (int*)alloc((size_t)BATCH * M1 * KNBR * 4);  // reused per stage
  int*   cnt  = (int*)alloc((size_t)BATCH * M1 * 4);         // reused per stage
  float* g    = (float*)alloc((size_t)BATCH * M4 * 1024 * 4);
  float* pooled = (float*)alloc((size_t)BATCH * 1024 * 4);
  _Float16* pooled16 = (_Float16*)alloc((size_t)BATCH * 1024 * 2);

  // ---- pack weights into WMMA B-fragment order (deterministic each call) --
  for (int i = 0; i < 16; ++i) {
    int tot = L[i].Kpad * L[i].Cout;
    pack_w_kernel<<<(tot + 255) / 256, 256, 0, stream>>>(
        (const float*)d_in[L[i].idx], wp[i], L[i].Cin, L[i].Cout, L[i].Kpad);
  }

  // ---- shared prep for one SA stage ----
  auto prep_sa = [&](const float* pos, int N, int M, float r2, int* idxb, float* ctrb) {
    fps_kernel<<<BATCH, 256, 0, stream>>>(pos, N, M, idxb);
    gather_ctr_kernel<<<(BATCH * M + 255) / 256, 256, 0, stream>>>(pos, idxb, ctrb, N, M);
    ball_query_kernel<<<(BATCH * M + 255) / 256, 256, 0, stream>>>(pos, ctrb, nbr, cnt, N, M, r2);
  };
  auto fill_sa = [&](SAArgs& a, const float* feat, const float* pos, int N, int M,
                     int wbase, int pbase, float* ctrb, float* featout) {
    a.feat = feat; a.pos = pos; a.ctr = ctrb; a.nbr = nbr; a.cnt = cnt;
    a.feat_out = featout; a.N = N; a.M = M;
    a.W0 = wp[wbase]; a.W1 = wp[wbase + 1]; a.W2 = wp[wbase + 2];
    a.b0 = (const float*)d_in[pbase + 1];  a.s0 = (const float*)d_in[pbase + 2];
    a.h0 = (const float*)d_in[pbase + 3];
    a.b1 = (const float*)d_in[pbase + 5];  a.s1 = (const float*)d_in[pbase + 6];
    a.h1 = (const float*)d_in[pbase + 7];
    a.b2 = (const float*)d_in[pbase + 9];  a.s2 = (const float*)d_in[pbase + 10];
    a.h2 = (const float*)d_in[pbase + 11];
  };

  SAArgs a;
  // Stage 1: N=4096 -> M=820, r=3, MLP [3,64,64,128]
  prep_sa(positions, 4096, M1, R2[0], idx1, ctr1);
  fill_sa(a, nullptr, positions, 4096, M1, 0, SA1, ctr1, feat1);
  sa_stage_kernel<0, 32, 64, 64, 64, 64, 128><<<BATCH * M1, 128, 0, stream>>>(a);
  // Stage 2: N=820 -> M=205, r=4, MLP [131,128,128,128]
  prep_sa(ctr1, M1, M2, R2[1], idx2, ctr2);
  fill_sa(a, feat1, ctr1, M1, M2, 3, SA2, ctr2, feat2);
  sa_stage_kernel<128, 160, 128, 128, 128, 128, 128><<<BATCH * M2, 128, 0, stream>>>(a);
  // Stage 3: N=205 -> M=52, r=5
  prep_sa(ctr2, M2, M3, R2[2], idx3, ctr3);
  fill_sa(a, feat2, ctr2, M2, M3, 6, SA3, ctr3, feat3);
  sa_stage_kernel<128, 160, 128, 128, 128, 128, 128><<<BATCH * M3, 128, 0, stream>>>(a);
  // Stage 4: N=52 -> M=13, r=7, MLP [131,128,128,256]
  prep_sa(ctr3, M3, M4, R2[3], idx4, ctr4);
  fill_sa(a, feat3, ctr3, M3, M4, 9, SA4, ctr4, feat4);
  sa_stage_kernel<128, 160, 128, 128, 128, 128, 256><<<BATCH * M4, 128, 0, stream>>>(a);

  // ---- global MLP + max pool + final linear ----
  GlobArgs ga;
  ga.feat = feat4; ga.pos = ctr4; ga.g = g; ga.R = BATCH * M4;
  ga.W0 = wp[12]; ga.W1 = wp[13]; ga.W2 = wp[14];
  ga.b0 = (const float*)d_in[GLB + 1];  ga.s0 = (const float*)d_in[GLB + 2];
  ga.h0 = (const float*)d_in[GLB + 3];
  ga.b1 = (const float*)d_in[GLB + 5];  ga.s1 = (const float*)d_in[GLB + 6];
  ga.h1 = (const float*)d_in[GLB + 7];
  ga.b2 = (const float*)d_in[GLB + 9];  ga.s2 = (const float*)d_in[GLB + 10];
  ga.h2 = (const float*)d_in[GLB + 11];
  glob_stage_kernel<<<(BATCH * M4 + 31) / 32, 128, 0, stream>>>(ga);

  pool_kernel<<<BATCH, 256, 0, stream>>>(g, pooled, pooled16, M4, 1024);

  final_linear_kernel<<<1, 128, 0, stream>>>(pooled16, wp[15],
                                             (const float*)d_in[LIN_B], (float*)d_out);
}
